// QRSparseLinear_4294967296362
// MI455X (gfx1250) — matrix operand order repaired and checked by reading, not compiled, not run on hardware
//
#include <hip/hip_runtime.h>
#include <hip/hip_bf16.h>

typedef unsigned int   u32;
typedef unsigned short u16t;

typedef __attribute__((ext_vector_type(8)))  float   v8f;
typedef __attribute__((ext_vector_type(8)))  __bf16  v8bf;
typedef __attribute__((ext_vector_type(16))) __bf16  v16bf;

#define B_     4
#define T_     2048
#define D_     4096
#define OUT_   4096
#define RANK_  128
#define SELECT_ 128
#define GROUP_ 16
#define ST_    204            // int(2048 * (1.0 - 0.9)) = 204
#define SCALE_CAP_ (448.0f * 6.0f)

// ---- CDNA5 async global->LDS support (guarded; fallback keeps same structure)
#if defined(__gfx1250__) && __has_builtin(__builtin_amdgcn_global_load_async_to_lds_b128)
#define HAVE_ASYNC_LDS 1
#else
#define HAVE_ASYNC_LDS 0
#endif

#if __has_builtin(__builtin_amdgcn_s_wait_asynccnt)
#define WAIT_ASYNC(n) __builtin_amdgcn_s_wait_asynccnt(n)
#elif HAVE_ASYNC_LDS
#define WAIT_ASYNC(n) asm volatile("s_wait_asynccnt %0" ::"i"(n) : "memory")
#else
#define WAIT_ASYNC(n) ((void)0)
#endif

#if HAVE_ASYNC_LDS
typedef int v4i __attribute__((vector_size(16)));
typedef __attribute__((address_space(1))) v4i* gptr4;
typedef __attribute__((address_space(3))) v4i* lptr4;
__device__ __forceinline__ void async_cp16(u16t* l, const u16t* g) {
    // global_load_async_to_lds_b128: 16B per lane, tracked by ASYNCcnt
    __builtin_amdgcn_global_load_async_to_lds_b128(
        (gptr4)const_cast<u16t*>(g), (lptr4)l, 0, 0);
}
#endif

__device__ __forceinline__ u16t f2bf(float f) {
    u32 u = __float_as_uint(f);
    u32 r = (u + 0x7FFFu + ((u >> 16) & 1u)) >> 16;   // round-to-nearest-even
    return (u16t)r;
}

__device__ __forceinline__ float fp4_round(float a) {
    float v = fabsf(a);
    float r;
    if      (v < 0.25f) r = 0.0f;
    else if (v < 0.75f) r = 0.5f;
    else if (v < 1.25f) r = 1.0f;
    else if (v < 1.75f) r = 1.5f;
    else if (v < 2.5f)  r = 2.0f;
    else if (v < 3.5f)  r = 3.0f;
    else if (v < 5.0f)  r = 4.0f;
    else                r = 6.0f;
    return copysignf(r, a);
}

// ---------------------------------------------------------------- init
__global__ void k_init(u32* wsU) {
    wsU[0] = 0u;
    wsU[1] = 0u;
}

// ---------------------------------------------------------------- scales
__global__ __launch_bounds__(256) void k_scales(
    const float* __restrict__ x, const float* __restrict__ scale_vec,
    const float* __restrict__ thr, u32* __restrict__ wsU)
{
    const size_t total = (size_t)B_ * T_ * D_;
    const float th = thr[0];
    float mp = 0.f, md = 0.f;
    for (size_t i = (size_t)blockIdx.x * blockDim.x + threadIdx.x; i < total;
         i += (size_t)gridDim.x * blockDim.x) {
        int d = (int)(i % D_);
        int t = (int)((i / D_) % T_);
        float xv = x[i];
        float av = fabsf(xv);
        if (t < ST_) {
            mp = fmaxf(mp, av);
        } else {
            float m = (fabsf(xv * scale_vec[d]) > th) ? av : 0.f;
            md = fmaxf(md, m);
        }
    }
    __shared__ float sp[256], sd[256];
    int tid = threadIdx.x;
    sp[tid] = mp; sd[tid] = md;
    __syncthreads();
    for (int s = 128; s > 0; s >>= 1) {
        if (tid < s) {
            sp[tid] = fmaxf(sp[tid], sp[tid + s]);
            sd[tid] = fmaxf(sd[tid], sd[tid + s]);
        }
        __syncthreads();
    }
    if (tid == 0) {
        atomicMax(&wsU[0], __float_as_uint(sp[0]));
        atomicMax(&wsU[1], __float_as_uint(sd[0]));
    }
}

// ---------------------------------------------------------------- pack W / vsT / utT (bf16)
__global__ __launch_bounds__(256) void k_pack(
    const float* __restrict__ W, const float* __restrict__ vs,
    const float* __restrict__ u_t, const int* __restrict__ rid,
    u16t* __restrict__ Wr, u16t* __restrict__ vsT, u16t* __restrict__ utT)
{
    const size_t stride = (size_t)gridDim.x * blockDim.x;
    const size_t t0 = (size_t)blockIdx.x * blockDim.x + threadIdx.x;
    for (size_t i = t0; i < (size_t)OUT_ * D_; i += stride) {
        size_t o = i / D_; int d = (int)(i % D_);
        Wr[i] = f2bf(W[o * D_ + rid[d]]);
    }
    for (size_t i = t0; i < (size_t)RANK_ * D_; i += stride) {
        size_t r = i / D_; size_t d = i % D_;
        vsT[i] = f2bf(vs[d * RANK_ + r]);          // vsT[r][d] = vs[d][r]
    }
    for (size_t i = t0; i < (size_t)OUT_ * RANK_; i += stride) {
        size_t o = i / RANK_; size_t r = i % RANK_;
        utT[i] = f2bf(u_t[r * OUT_ + o]);          // utT[o][r] = u_t[r][o]
    }
}

// ---------------------------------------------------------------- quantize (one block per token)
__global__ __launch_bounds__(256) void k_quant(
    const float* __restrict__ x, const int* __restrict__ rid,
    const float* __restrict__ scale_vec, const float* __restrict__ thr,
    const u32* __restrict__ wsU, u16t* __restrict__ Xq, u16t* __restrict__ LRin)
{
    const int m = blockIdx.x;            // token index in [0, B*T)
    const int t = m % T_;
    const bool prefill = (t < ST_);
    const float smax = __uint_as_float(wsU[prefill ? 0 : 1]);
    const float s = fmaxf(smax / SCALE_CAP_, 1e-12f);
    const float inv_s = 1.0f / s;
    const float th = thr[0];
    const int g = threadIdx.x;           // group id, 16 elems each, 256 groups = D
    const size_t base = (size_t)m * D_;

    float vals[GROUP_];
    float gmax = 0.f;
    #pragma unroll
    for (int i = 0; i < GROUP_; i++) {
        int d = g * GROUP_ + i;
        int src = rid[d];
        float xv = x[base + src];
        float lr = 0.f;
        if (!prefill) {
            bool msk = fabsf(xv * scale_vec[src]) > th;
            lr = msk ? 0.f : xv;
            xv = msk ? xv : 0.f;
        }
        LRin[base + src] = f2bf(lr);
        float xr = xv * inv_s;
        vals[i] = xr;
        gmax = fmaxf(gmax, fabsf(xr));
    }
    if (g < (SELECT_ / GROUP_)) {        // "hi" region: exact
        #pragma unroll
        for (int i = 0; i < GROUP_; i++)
            Xq[base + g * GROUP_ + i] = f2bf(vals[i] * s);
    } else {                             // fp4 quant-dequant per group of 16
        float gs = (gmax > 0.f) ? gmax * (1.0f / 6.0f) : 1.0f;
        float inv_gs = 1.0f / gs;
        #pragma unroll
        for (int i = 0; i < GROUP_; i++) {
            float q = fp4_round(vals[i] * inv_gs) * gs;
            Xq[base + g * GROUP_ + i] = f2bf(q * s);
        }
    }
}

// ---------------------------------------------------------------- WMMA GEMM
// C[m][n] = sum_k A[m][k] * Bm[n][k]   (A: MxK bf16 row-major, Bm: NxK bf16 row-major)
// MODE 0: Cf = gemm + bias;  MODE 1: Cb(bf16) = gemm;  MODE 2: Cf += gemm
// Double-buffered LDS tiles; on CDNA5 uses GLOBAL_LOAD_ASYNC_TO_LDS_B128 so the
// next K-tile streams into LDS while the current tile's 16 WMMAs execute.
template <int MODE>
__global__ __launch_bounds__(256) void k_gemm(
    const u16t* __restrict__ A, const u16t* __restrict__ Bm,
    const float* __restrict__ bias, float* __restrict__ Cf,
    u16t* __restrict__ Cb, int M, int N, int K)
{
    constexpr int TM = 128, TN = 128, TK = 64;
    __shared__ __attribute__((aligned(32))) u16t As[2][TM * TK];
    __shared__ __attribute__((aligned(32))) u16t Bs[2][TN * TK];

    const int nb = N / TN;
    const int m0 = (blockIdx.x / nb) * TM;
    const int n0 = (blockIdx.x % nb) * TN;
    const int tid  = threadIdx.x;
    const int lane = tid & 31;
    const int half = lane >> 4;
    const int ln   = lane & 15;
    const int w    = tid >> 5;
    const int wm = (w & 3) * 32;     // 4 waves along M, 32 rows each
    const int wn = (w >> 2) * 64;    // 2 waves along N, 64 cols each

    // Per-thread tile chunks: 8 x 16B (4 for A, 4 for B) per K-step.
    auto issue_tile = [&](int buf, int k0) {
        #pragma unroll
        for (int j = 0; j < 4; j++) {
            int c = tid + j * 256;               // chunk id, 1024 chunks of 8 bf16
            int row = c >> 3, col = (c & 7) * 8;
            const u16t* g = &A[(size_t)(m0 + row) * K + k0 + col];
            u16t* l = &As[buf][row * TK + col];
#if HAVE_ASYNC_LDS
            async_cp16(l, g);
#else
            *(uint4*)l = *(const uint4*)g;
#endif
        }
        #pragma unroll
        for (int j = 0; j < 4; j++) {
            int c = tid + j * 256;
            int row = c >> 3, col = (c & 7) * 8;
            const u16t* g = &Bm[(size_t)(n0 + row) * K + k0 + col];
            u16t* l = &Bs[buf][row * TK + col];
#if HAVE_ASYNC_LDS
            async_cp16(l, g);
#else
            *(uint4*)l = *(const uint4*)g;
#endif
        }
    };

    v8f acc[2][4];
    #pragma unroll
    for (int sm = 0; sm < 2; sm++)
        #pragma unroll
        for (int sn = 0; sn < 4; sn++)
            acc[sm][sn] = v8f{0.f,0.f,0.f,0.f,0.f,0.f,0.f,0.f};

    const int nIter = K / TK;
    issue_tile(0, 0);

    for (int it = 0; it < nIter; it++) {
        const int cur = it & 1;
        if (it + 1 < nIter) {
            issue_tile(cur ^ 1, (it + 1) * TK);   // prefetch next tile into other buffer
            WAIT_ASYNC(8);                         // previous tile's 8 loads complete (in-order)
        } else {
            WAIT_ASYNC(0);
        }
        __syncthreads();

        const u16t* Asb = As[cur];
        const u16t* Bsb = Bs[cur];
        #pragma unroll
        for (int kk = 0; kk < TK; kk += 32) {
            v16bf afrag[2], bfrag[4];
            #pragma unroll
            for (int sm = 0; sm < 2; sm++) {
                int row = wm + sm * 16 + ln;
                // A 16x32 bf16 lane layout: half 0 -> K {0..7,16..23}, half 1 -> K {8..15,24..31}
                v8bf lo = *reinterpret_cast<const v8bf*>(&Asb[row * TK + kk + half * 8]);
                v8bf hi = *reinterpret_cast<const v8bf*>(&Asb[row * TK + kk + 16 + half * 8]);
                afrag[sm] = __builtin_shufflevector(lo, hi,
                    0,1,2,3,4,5,6,7,8,9,10,11,12,13,14,15);
            }
            #pragma unroll
            for (int sn = 0; sn < 4; sn++) {
                int col = wn + sn * 16 + ln;
                // B 32x16 bf16 lane layout: half 0 -> K {0..7,16..23}, half 1 -> K {8..15,24..31}
                v8bf blo = *reinterpret_cast<const v8bf*>(&Bsb[col * TK + kk + half * 8]);
                v8bf bhi = *reinterpret_cast<const v8bf*>(&Bsb[col * TK + kk + 16 + half * 8]);
                bfrag[sn] = __builtin_shufflevector(blo, bhi,
                    0,1,2,3,4,5,6,7,8,9,10,11,12,13,14,15);
            }
            #pragma unroll
            for (int sm = 0; sm < 2; sm++)
                #pragma unroll
                for (int sn = 0; sn < 4; sn++)
                    acc[sm][sn] = __builtin_amdgcn_wmma_f32_16x16x32_bf16(
                        false, afrag[sm], false, bfrag[sn],
                        (short)0, acc[sm][sn], false, false);
        }
        __syncthreads();   // all waves done with 'cur' before it is refilled
    }

    #pragma unroll
    for (int sm = 0; sm < 2; sm++) {
        #pragma unroll
        for (int sn = 0; sn < 4; sn++) {
            int n = n0 + wn + sn * 16 + ln;
            float bv = (MODE == 0) ? bias[n] : 0.f;
            #pragma unroll
            for (int r = 0; r < 8; r++) {
                int mrow = m0 + wm + sm * 16 + half * 8 + r;
                size_t idx = (size_t)mrow * N + n;
                float v = acc[sm][sn][r] + bv;
                if      (MODE == 1) Cb[idx] = f2bf(v);
                else if (MODE == 2) Cf[idx] += v;
                else                Cf[idx] = v;
            }
        }
    }
}

// ---------------------------------------------------------------- launch
extern "C" void kernel_launch(void* const* d_in, const int* in_sizes, int n_in,
                              void* d_out, int out_size, void* d_ws, size_t ws_size,
                              hipStream_t stream) {
    const float* x         = (const float*)d_in[0];
    const float* W         = (const float*)d_in[1];
    const float* bias      = (const float*)d_in[2];
    const float* vs        = (const float*)d_in[3];
    const float* u_t       = (const float*)d_in[4];
    const float* scale_vec = (const float*)d_in[5];
    const float* thr       = (const float*)d_in[6];
    const int*   rid       = (const int*)d_in[7];
    float* out = (float*)d_out;

    // workspace layout
    char* ws = (char*)d_ws;
    size_t off = 0;
    u32*  wsU  = (u32*)(ws + off);  off += 256;
    u16t* Xq   = (u16t*)(ws + off); off += (size_t)B_ * T_ * D_ * 2;     // 64 MB
    u16t* Wr   = (u16t*)(ws + off); off += (size_t)OUT_ * D_ * 2;        // 32 MB
    u16t* LRin = (u16t*)(ws + off); off += (size_t)B_ * T_ * D_ * 2;     // 64 MB
    u16t* vsT  = (u16t*)(ws + off); off += (size_t)RANK_ * D_ * 2;       //  1 MB
    u16t* utT  = (u16t*)(ws + off); off += (size_t)OUT_ * RANK_ * 2;     //  1 MB
    u16t* tmp  = (u16t*)(ws + off); off += (size_t)B_ * T_ * RANK_ * 2;  //  2 MB

    const int M = B_ * T_;   // 8192

    k_init<<<1, 1, 0, stream>>>(wsU);
    k_scales<<<2048, 256, 0, stream>>>(x, scale_vec, thr, wsU);
    k_pack<<<4096, 256, 0, stream>>>(W, vs, u_t, rid, Wr, vsT, utT);
    k_quant<<<M, 256, 0, stream>>>(x, rid, scale_vec, thr, wsU, Xq, LRin);

    // main dense GEMM: out = Xq @ Wr^T + bias      (M x OUT, K = D)
    k_gemm<0><<<(M / 128) * (OUT_ / 128), 256, 0, stream>>>(
        Xq, Wr, bias, out, nullptr, M, OUT_, D_);
    // low-rank stage 1: tmp = LRin @ vs            (M x RANK, K = D), bf16 out
    k_gemm<1><<<(M / 128) * (RANK_ / 128), 256, 0, stream>>>(
        LRin, vsT, nullptr, nullptr, tmp, M, RANK_, D_);
    // low-rank stage 2: out += tmp @ u_t           (M x OUT, K = RANK)
    k_gemm<2><<<(M / 128) * (OUT_ / 128), 256, 0, stream>>>(
        tmp, utT, nullptr, out, nullptr, M, OUT_, RANK_);
}
